// SemiFullGN_28621662061209
// MI455X (gfx1250) — matrix operand order repaired and checked by reading, not compiled, not run on hardware
//
#include <hip/hip_runtime.h>
#include <hip/hip_bf16.h>

// ============================================================================
// SemiFullGN forward for MI455X (gfx1250, wave32, WMMA).
// All GEMM-shaped compute runs on v_wmma_f32_16x16x32_bf16 (bf16 in, f32 acc).
// Each wave computes a 16x64 output tile (4 accumulators) so the A panel is
// loaded once per K-chunk and reused by 4 WMMAs. B fragments are single
// contiguous 32-byte loads. No guards in the hot loop: M%16==0, Nc%64==0,
// K%32==0 by construction (zero-padded). Convs use channels-last [N,L,C] with
// halo rows so im2col is a contiguous sliding window.
// ============================================================================

typedef __bf16 bf16;
typedef __attribute__((ext_vector_type(16))) __bf16 v16bf;
typedef __attribute__((ext_vector_type(8)))  __bf16 v8bf;
typedef __attribute__((ext_vector_type(8)))  float  v8f;

#define DEVFN static __device__ __forceinline__

DEVFN v16bf join16(v8bf lo, v8bf hi) {
  return __builtin_shufflevector(lo, hi, 0, 1, 2, 3, 4, 5, 6, 7, 8, 9, 10, 11,
                                 12, 13, 14, 15);
}

// ---------------------------------------------------------------------------
// Generic GEMM: out = act(A[M,K] @ Wt[Nc,K]^T + bias), 16x64 tile per wave.
// Requirements: M%16==0, Nc%64==0, K%32==0, lda%8==0, 16B-aligned buffers.
// Operand k order used by this code (contiguous, same for A and B):
//   A lane(0-15)=row, K run {0..15}; lanes 16-31: K {16..31}
//   B lane(0-15)=col, K run {0..15}; lanes 16-31: K {16..31}
// ---------------------------------------------------------------------------
__global__ void wmma_gemm_kernel(const bf16* __restrict__ A, int lda,
                                 const bf16* __restrict__ Wt,
                                 const float* __restrict__ bias,
                                 long M, int K, int Nc,
                                 float* __restrict__ outF, int ldf,
                                 bf16* __restrict__ outB, int ldbo, int act) {
  const int lane = threadIdx.x & 31;
  const int wave = threadIdx.x >> 5;
  const int tilesN = Nc >> 6;  // groups of 64 output columns
  const long tile = (long)blockIdx.x * (blockDim.x >> 5) + wave;
  if (tile >= (M >> 4) * (long)tilesN) return;
  const long m0 = (tile / tilesN) << 4;
  const int  n0 = (int)(tile % tilesN) << 6;

  const int half = (lane < 16) ? 0 : 1;
  const bf16* pa = A + (size_t)(m0 + (lane & 15)) * lda + half * 16;
  const bf16* pb = Wt + (size_t)(n0 + (lane & 15)) * K + half * 16;
  const size_t bstride = (size_t)16 * K;  // 16 columns per sub-tile
  __builtin_prefetch(pb, 0, 1);

  v8f acc[4];
#pragma unroll
  for (int t = 0; t < 4; ++t) acc[t] = (v8f){0.f, 0.f, 0.f, 0.f,
                                             0.f, 0.f, 0.f, 0.f};
  for (int k0 = 0; k0 < K; k0 += 32) {
    const v8bf a0 = *(const v8bf*)(pa + k0);
    const v8bf a1 = *(const v8bf*)(pa + k0 + 8);
    const v16bf av = join16(a0, a1);
#pragma unroll
    for (int t = 0; t < 4; ++t) {
      const v16bf bv = *(const v16bf*)(pb + bstride * t + k0);
      acc[t] = __builtin_amdgcn_wmma_f32_16x16x32_bf16(
          false, av, false, bv, (short)0, acc[t], false, false);
    }
  }

#pragma unroll
  for (int t = 0; t < 4; ++t) {
    const int col = n0 + 16 * t + (lane & 15);
    const float bb = bias ? bias[col] : 0.0f;
#pragma unroll
    for (int v = 0; v < 8; ++v) {
      const long row = m0 + ((lane < 16) ? v : (v + 8));
      float y = acc[t][v] + bb;
      if (act) y = (y > 0.0f) ? y : 0.2f * y;
      if (outF) outF[(size_t)row * ldf + col] = y;
      if (outB) outB[(size_t)row * ldbo + col] = (bf16)y;
    }
  }
}

// ---------------------------------------------------------------------------
// Conv1d as GEMM over channels-last x [N, Lalloc, Cin] (halo rows hold zeros;
// x points at logical row l=-pad). K = kw*Cin ordered (j, cin): the receptive
// field of output row (n,p) is the contiguous window at (n*Lalloc+p)*Cin.
// Output channels-last [N, outLalloc, Cout] at row offset outPad; outF
// (optional) is plain rows [(n*Lout+p)*Cout + c] for BN. 16x64 tile per wave.
// ---------------------------------------------------------------------------
__global__ void wmma_conv_kernel(const bf16* __restrict__ x, int Cin,
                                 int Lalloc,
                                 const bf16* __restrict__ Wt,
                                 const float* __restrict__ bias,
                                 int N, int Lout, int K, int Cout,
                                 float* __restrict__ outF,
                                 bf16* __restrict__ outB, int outLalloc,
                                 int outPad, int act) {
  const int lane = threadIdx.x & 31;
  const int wave = threadIdx.x >> 5;
  const long M = (long)N * Lout;
  const int tilesN = Cout >> 6;
  const long tile = (long)blockIdx.x * (blockDim.x >> 5) + wave;
  if (tile >= (M >> 4) * (long)tilesN) return;
  const long m0 = (tile / tilesN) << 4;
  const int  n0 = (int)(tile % tilesN) << 6;

  const int half = (lane < 16) ? 0 : 1;
  const long r = m0 + (lane & 15);
  const int nn = (int)(r / Lout);
  const int pp = (int)(r % Lout);
  const bf16* pa = x + ((size_t)nn * Lalloc + pp) * Cin + half * 16;
  const bf16* pb = Wt + (size_t)(n0 + (lane & 15)) * K + half * 16;
  const size_t bstride = (size_t)16 * K;
  __builtin_prefetch(pb, 0, 1);

  v8f acc[4];
#pragma unroll
  for (int t = 0; t < 4; ++t) acc[t] = (v8f){0.f, 0.f, 0.f, 0.f,
                                             0.f, 0.f, 0.f, 0.f};
  for (int k0 = 0; k0 < K; k0 += 32) {
    const v8bf a0 = *(const v8bf*)(pa + k0);
    const v8bf a1 = *(const v8bf*)(pa + k0 + 8);
    const v16bf av = join16(a0, a1);
#pragma unroll
    for (int t = 0; t < 4; ++t) {
      const v16bf bv = *(const v16bf*)(pb + bstride * t + k0);
      acc[t] = __builtin_amdgcn_wmma_f32_16x16x32_bf16(
          false, av, false, bv, (short)0, acc[t], false, false);
    }
  }

#pragma unroll
  for (int t = 0; t < 4; ++t) {
    const int col = n0 + 16 * t + (lane & 15);
    const float bb = bias ? bias[col] : 0.0f;
#pragma unroll
    for (int v = 0; v < 8; ++v) {
      const long row = m0 + ((lane < 16) ? v : (v + 8));
      const int n2 = (int)(row / Lout);
      const int p2 = (int)(row % Lout);
      float y = acc[t][v] + bb;
      if (act) y = (y > 0.0f) ? y : 0.2f * y;
      if (outF) outF[(size_t)row * Cout + col] = y;
      if (outB)
        outB[((size_t)n2 * outLalloc + p2 + outPad) * Cout + col] = (bf16)y;
    }
  }
}

// ---------------------------------------------------------------------------
// Packing / casting kernels
// ---------------------------------------------------------------------------
// Linear: wt[Nc, Kp] = cast(w[K, Nc])^T, zero pad K..Kp.
__global__ void pack_wt_kernel(const float* __restrict__ w,
                               bf16* __restrict__ wt, int K, int Kp, int Nc) {
  long i = (long)blockIdx.x * blockDim.x + threadIdx.x;
  if (i >= (long)Kp * Nc) return;
  const int n = (int)(i / Kp);
  const int k = (int)(i % Kp);
  wt[i] = (k < K) ? (bf16)w[(size_t)k * Nc + n] : (bf16)0.0f;
}

// Conv: wt[Cout, j*Cin+cin] = cast(w[Cout, cin, j])  (K = Cin*kw, %32==0)
__global__ void pack_conv_wt_kernel(const float* __restrict__ w,
                                    bf16* __restrict__ wt, int Cout, int Cin,
                                    int kw) {
  const int K = Cin * kw;
  long i = (long)blockIdx.x * blockDim.x + threadIdx.x;
  if (i >= (long)Cout * K) return;
  const int co = (int)(i / K);
  const int k = (int)(i % K);
  const int j = k / Cin;
  const int ci = k % Cin;
  wt[i] = (bf16)w[(size_t)co * K + (size_t)ci * kw + j];
}

// dst[rows, Wp] = cast(src[rows, W]), zero pad W..Wp.
__global__ void cast_pad_kernel(const float* __restrict__ src,
                                bf16* __restrict__ dst, long rows, int W,
                                int Wp) {
  long i = (long)blockIdx.x * blockDim.x + threadIdx.x;
  if (i >= rows * Wp) return;
  const long r = i / Wp;
  const int c = (int)(i % Wp);
  dst[i] = (c < W) ? (bf16)src[r * W + c] : (bf16)0.0f;
}

// Zero halo rows l=0 and l=Lalloc-1 of channels-last padded buffer.
__global__ void zero_halo_kernel(bf16* __restrict__ buf, long N, int Lalloc,
                                 int C) {
  long i = (long)blockIdx.x * blockDim.x + threadIdx.x;
  if (i >= N * 2 * C) return;
  const long n = i / (2 * C);
  const int rem = (int)(i % (2 * C));
  const int c = rem % C;
  const int l = (rem < C) ? 0 : (Lalloc - 1);
  buf[((size_t)n * Lalloc + l) * C + c] = (bf16)0.0f;
}

// Edge concat (8-byte vector copies): Xe[e] = [atomB[i1] | atomB[i2] | nbrB[e]]
__global__ void edge_concat_kernel(const bf16* __restrict__ atomB,
                                   const bf16* __restrict__ nbrB,
                                   const int* __restrict__ idx1,
                                   const int* __restrict__ idx2,
                                   bf16* __restrict__ Xe, long E) {
  long i = (long)blockIdx.x * blockDim.x + threadIdx.x;  // E * 16 quads
  if (i >= E * 16) return;
  const long e = i >> 4;
  const int q = (int)(i & 15);  // 4 bf16 per quad
  const uint2* a1 = (const uint2*)(atomB + (size_t)idx1[e] * 64) + q;
  const uint2* a2 = (const uint2*)(atomB + (size_t)idx2[e] * 64) + q;
  const uint2* nb = (const uint2*)(nbrB + e * 64) + q;
  uint2* row = (uint2*)(Xe + e * 192);
  row[q] = *a1;
  row[16 + q] = *a2;
  row[32 + q] = *nb;
}

// rho[idx1[e], c] += src[e, c] / num_nbrs[idx1[e]]
__global__ void scatter_scaled_kernel(const float* __restrict__ src,
                                      const int* __restrict__ idx1,
                                      const float* __restrict__ num_nbrs,
                                      float* __restrict__ rho, long E) {
  long i = (long)blockIdx.x * blockDim.x + threadIdx.x;
  if (i >= E * 64) return;
  const long e = i >> 6;
  const int c = (int)(i & 63);
  const int a = idx1[e];
  const float inv = 1.0f / num_nbrs[a];
  atomicAdd(&rho[(size_t)a * 64 + c], src[i] * inv);
}

// out[N,2C] bf16 = concat(a[N,C], b[N,C]) (both f32)
__global__ void concat2_cast_kernel(const float* __restrict__ a,
                                    const float* __restrict__ b,
                                    bf16* __restrict__ out, long N, int C) {
  long i = (long)blockIdx.x * blockDim.x + threadIdx.x;
  if (i >= N * 2 * C) return;
  const long n = i / (2 * C);
  const int c = (int)(i % (2 * C));
  const float v = (c < C) ? a[n * C + c] : b[n * C + (c - C)];
  out[i] = (bf16)v;
}

// accF += delta; accB = bf16(accF)
__global__ void add_cast_kernel(float* __restrict__ accF,
                                bf16* __restrict__ accB,
                                const float* __restrict__ delta, long n) {
  long i = (long)blockIdx.x * blockDim.x + threadIdx.x;
  if (i >= n) return;
  const float v = accF[i] + delta[i];
  accF[i] = v;
  accB[i] = (bf16)v;
}

// BN statistics over rows of [M, C]: sums[0:C]=sum, sums[C:2C]=sumsq.
__global__ void bn_stats_rows_kernel(const float* __restrict__ x, long M,
                                     int C, float* __restrict__ sums) {
  extern __shared__ float sm[];
  for (int i = threadIdx.x; i < 2 * C; i += blockDim.x) sm[i] = 0.0f;
  __syncthreads();
  const long total = M * C;
  for (long i = (long)blockIdx.x * blockDim.x + threadIdx.x; i < total;
       i += (long)gridDim.x * blockDim.x) {
    const int c = (int)(i % C);
    const float v = x[i];
    atomicAdd(&sm[c], v);
    atomicAdd(&sm[C + c], v * v);
  }
  __syncthreads();
  for (int i = threadIdx.x; i < 2 * C; i += blockDim.x)
    atomicAdd(&sums[i], sm[i]);
}

__global__ void bn_finalize_kernel(const float* __restrict__ sums,
                                   float* __restrict__ stats, int C,
                                   float count) {
  int c = blockIdx.x * blockDim.x + threadIdx.x;
  if (c >= C) return;
  const float m = sums[c] / count;
  const float var = sums[C + c] / count - m * m;
  stats[c] = m;
  stats[C + c] = rsqrtf(var + 1e-5f);
}

// atom += bn(vi); refresh bf16 copy.
__global__ void bn_res_update_kernel(float* __restrict__ atomF,
                                     bf16* __restrict__ atomB,
                                     const float* __restrict__ vi,
                                     const float* __restrict__ stats,
                                     const float* __restrict__ g,
                                     const float* __restrict__ b, long N,
                                     int C) {
  long i = (long)blockIdx.x * blockDim.x + threadIdx.x;
  if (i >= N * C) return;
  const int c = (int)(i % C);
  const float xn = (vi[i] - stats[c]) * stats[C + c];
  const float v = atomF[i] + xn * g[c] + b[c];
  atomF[i] = v;
  atomB[i] = (bf16)v;
}

// bf16(lrelu(bn(x))) for x as rows [N*Lrow, C]; output written channels-last
// into padded buffer rows (n*Lalloc + p + pad).
__global__ void bn_lrelu_cast_pad_kernel(const float* __restrict__ x,
                                         const float* __restrict__ stats,
                                         const float* __restrict__ g,
                                         const float* __restrict__ b,
                                         bf16* __restrict__ out, long N,
                                         int Lrow, int C, int Lalloc,
                                         int pad) {
  long i = (long)blockIdx.x * blockDim.x + threadIdx.x;
  if (i >= N * Lrow * C) return;
  const int c = (int)(i % C);
  const long m = i / C;
  const long n = m / Lrow;
  const int p = (int)(m % Lrow);
  float y = (x[i] - stats[c]) * stats[C + c] * g[c] + b[c];
  y = (y > 0.0f) ? y : 0.2f * y;
  out[((size_t)n * Lalloc + p + pad) * C + c] = (bf16)y;
}

// pd [N,512] --BN+lrelu--> x0 channels-last [N,8,64]: x0[n,l,c]=f(pd[n,c*8+l])
__global__ void pd_to_x0_kernel(const float* __restrict__ pd,
                                const float* __restrict__ stats,
                                const float* __restrict__ g,
                                const float* __restrict__ b,
                                bf16* __restrict__ x0, long N) {
  long i = (long)blockIdx.x * blockDim.x + threadIdx.x;
  if (i >= N * 512) return;
  const long n = i / 512;
  const int idx = (int)(i % 512);
  const int c = idx >> 3;
  const int l = idx & 7;
  float y = (pd[i] - stats[idx]) * stats[512 + idx] * g[idx] + b[idx];
  y = (y > 0.0f) ? y : 0.2f * y;
  x0[((size_t)n * 8 + l) * 64 + c] = (bf16)y;
}

// Per-atom head gather: urB[N,256]=[u|r], fdB[N,128]=u-r, cellB[N,32] (9+pad0)
__global__ void head_gather_kernel(const int* __restrict__ crys,
                                   const float* __restrict__ u,
                                   const float* __restrict__ r,
                                   const float* __restrict__ cell,
                                   bf16* __restrict__ urB,
                                   bf16* __restrict__ fdB,
                                   bf16* __restrict__ cellB, int N) {
  int n = blockIdx.x * blockDim.x + threadIdx.x;
  if (n >= N) return;
  const int ci = crys[n];
  const float* up = u + (size_t)ci * 128;
  const float* rp = r + (size_t)ci * 128;
  bf16* o = urB + (size_t)n * 256;
  bf16* f = fdB + (size_t)n * 128;
  for (int j = 0; j < 128; ++j) {
    const float uv = up[j], rv = rp[j];
    o[j] = (bf16)uv;
    o[128 + j] = (bf16)rv;
    f[j] = (bf16)(uv - rv);
  }
  const float* cp = cell + (size_t)ci * 9;
  bf16* cb = cellB + (size_t)n * 32;
  for (int j = 0; j < 32; ++j) cb[j] = (j < 9) ? (bf16)cp[j] : (bf16)0.0f;
}

// Final conv c5: x4 channels-last [N,4,256], w [1,256,2] -> out [N,3] f32.
__global__ void c5_kernel(const bf16* __restrict__ x4,
                          const float* __restrict__ w,
                          const float* __restrict__ b,
                          float* __restrict__ out, int N) {
  long i = (long)blockIdx.x * blockDim.x + threadIdx.x;
  if (i >= (long)N * 3) return;
  const int n = (int)(i / 3);
  const int p = (int)(i % 3);
  const bf16* xr = x4 + (size_t)n * 4 * 256;
  float s = b[0];
  for (int c = 0; c < 256; ++c) {
    s += w[c * 2 + 0] * (float)xr[(p + 0) * 256 + c] +
         w[c * 2 + 1] * (float)xr[(p + 1) * 256 + c];
  }
  out[i] = s;
}

// ---------------------------------------------------------------------------
// Host-side helpers
// ---------------------------------------------------------------------------
static void gemm(hipStream_t s, const bf16* A, int lda, const bf16* Wt,
                 const float* bias, long M, int K, int Nc, float* outF,
                 int ldf, bf16* outB, int ldbo, int act) {
  const long tiles = (M >> 4) * (long)(Nc >> 6);
  const long blocks = (tiles + 7) / 8;
  wmma_gemm_kernel<<<(unsigned)blocks, 256, 0, s>>>(A, lda, Wt, bias, M, K, Nc,
                                                    outF, ldf, outB, ldbo, act);
}

static void conv(hipStream_t s, const bf16* x, int Cin, int Lalloc,
                 const bf16* Wt, const float* bias, long N, int Lout, int kw,
                 int Cout, float* outF, bf16* outB, int outLalloc, int outPad,
                 int act) {
  const long M = N * Lout;
  const int K = Cin * kw;
  const long tiles = (M >> 4) * (long)(Cout >> 6);
  const long blocks = (tiles + 7) / 8;
  wmma_conv_kernel<<<(unsigned)blocks, 256, 0, s>>>(
      x, Cin, Lalloc, Wt, bias, (int)N, Lout, K, Cout, outF, outB, outLalloc,
      outPad, act);
}

static void pack_wt(hipStream_t s, const float* w, bf16* wt, int K, int Kp,
                    int Nc) {
  const long n = (long)Kp * Nc;
  pack_wt_kernel<<<(unsigned)((n + 255) / 256), 256, 0, s>>>(w, wt, K, Kp, Nc);
}

static void pack_cw(hipStream_t s, const float* w, bf16* wt, int Cout, int Cin,
                    int kw) {
  const long n = (long)Cout * Cin * kw;
  pack_conv_wt_kernel<<<(unsigned)((n + 255) / 256), 256, 0, s>>>(w, wt, Cout,
                                                                  Cin, kw);
}

// Input index map (setup_inputs order; params flattened as a jax pytree:
// dict keys sorted, lists in order, each {"b","w"} pair sorted b-then-w).
enum {
  I_ATOM = 0, I_NBR, I_IDX1, I_IDX2, I_NUMN, I_CRYS, I_U, I_R, I_CELL,
  P_ANF_B = 9, P_ANF_W,
  P_C1_B, P_C1_W, P_C1_BNB, P_C1_BNG,
  P_C2_B, P_C2_W, P_C2_BNB, P_C2_BNG,
  P_C3_B, P_C3_W, P_C4_B, P_C4_W, P_C5_B, P_C5_W,
  P_CELL_B, P_CELL_W,
  P_CONV0 = 27,  // 3 layers x 14: bn_b, bn_g, pe0_b, pe0_w, pe1_b, pe1_w,
                 // pe2_b, pe2_w, pv0_b, pv0_w, pv1_b, pv1_w, pv2_b, pv2_w
  P_EDGE_B = 69, P_EDGE_W,
  P_FD_B = 71, P_FD_W, P_FEAT_B, P_FEAT_W, P_NODE_B, P_NODE_W,
  P_PP_B = 77, P_PP_W, P_PP_BNB, P_PP_BNG
};

extern "C" void kernel_launch(void* const* d_in, const int* in_sizes, int n_in,
                              void* d_out, int out_size, void* d_ws,
                              size_t ws_size, hipStream_t stream) {
  (void)in_sizes; (void)n_in; (void)out_size; (void)ws_size;
  const long NA = 60000, NE = 720000;

  const float* atom_fea = (const float*)d_in[I_ATOM];
  const float* nbr_fea  = (const float*)d_in[I_NBR];
  const int*   idx1     = (const int*)d_in[I_IDX1];
  const int*   idx2     = (const int*)d_in[I_IDX2];
  const float* num_nbrs = (const float*)d_in[I_NUMN];
  const int*   crys     = (const int*)d_in[I_CRYS];
  const float* u_feat   = (const float*)d_in[I_U];
  const float* r_feat   = (const float*)d_in[I_R];
  const float* cell     = (const float*)d_in[I_CELL];
  float* out = (float*)d_out;

  // ---- workspace carving ----
  char* base = (char*)d_ws;
  size_t off = 0;
  auto alloc = [&](size_t bytes) -> void* {
    off = (off + 255) & ~(size_t)255;
    void* p = base + off;
    off += bytes;
    return p;
  };
#define ALLOC_F(n) (float*)alloc((size_t)(n) * 4)
#define ALLOC_B(n) (bf16*)alloc((size_t)(n) * 2)

  // packed weights (bf16)
  bf16* node_wt = ALLOC_B(64 * 96);
  bf16* edge_wt = ALLOC_B(64 * 64);
  bf16* pe_wt[3][3];
  bf16* pv_wt[3][3];
  for (int l = 0; l < 3; ++l) {
    pe_wt[l][0] = ALLOC_B(64 * 192);
    pe_wt[l][1] = ALLOC_B(64 * 64);
    pe_wt[l][2] = ALLOC_B(64 * 64);
    pv_wt[l][0] = ALLOC_B(64 * 128);
    pv_wt[l][1] = ALLOC_B(64 * 64);
    pv_wt[l][2] = ALLOC_B(64 * 64);
  }
  bf16* feat_wt = ALLOC_B(512 * 256);
  bf16* fd_wt   = ALLOC_B(128 * 128);
  bf16* anf_wt  = ALLOC_B(128 * 128);
  bf16* cell_wt = ALLOC_B(128 * 32);
  bf16* pp_wt   = ALLOC_B(512 * 896);
  bf16* c1_wt   = ALLOC_B(512 * 192);
  bf16* c2_wt   = ALLOC_B(512 * 1536);
  bf16* c3_wt   = ALLOC_B(256 * 1536);
  bf16* c4_wt   = ALLOC_B(256 * 768);

  // activations
  bf16* afB    = ALLOC_B(NA * 96);          // atom_fea, K pad 92->96
  bf16* nfB    = ALLOC_B(NE * 64);          // nbr_fea,  K pad 41->64
  float* atomF = ALLOC_F(NA * 64);
  bf16*  atomB = ALLOC_B(NA * 64);
  float* nbrF  = ALLOC_F(NE * 64);
  bf16*  nbrB  = ALLOC_B(NE * 64);
  bf16*  Xe    = ALLOC_B(NE * 192);
  bf16*  h1B   = ALLOC_B(NE * 64);
  bf16*  h2B   = ALLOC_B(NE * 64);
  float* ekF   = ALLOC_F(NE * 64);
  float* rho   = ALLOC_F(NA * 64);
  bf16*  avB   = ALLOC_B(NA * 128);
  bf16*  hv1B  = ALLOC_B(NA * 64);
  bf16*  hv2B  = ALLOC_B(NA * 64);
  float* viF   = ALLOC_F(NA * 64);
  float* eksum = ALLOC_F(NA * 64);
  bf16*  anfInB = ALLOC_B(NA * 128);
  bf16*  urB   = ALLOC_B(NA * 256);
  bf16*  fdB   = ALLOC_B(NA * 128);
  bf16*  cellB = ALLOC_B(NA * 32);          // K pad 9->32
  bf16*  ffB   = ALLOC_B(NA * 896);
  float* pdF   = ALLOC_F(NA * 512);
  bf16*  x0B   = ALLOC_B(NA * 8 * 64);      // channels-last [N,8,64]
  float* y1F   = ALLOC_F(NA * 6 * 512);     // rows [(n*6+p), 512]
  bf16*  x1B   = ALLOC_B(NA * 6 * 512);     // CL [N,6,512], no halo
  float* y2F   = ALLOC_F(NA * 4 * 512);
  bf16*  x2B   = ALLOC_B(NA * 6 * 512);     // CL [N,6,512], halo pad=1
  bf16*  x3B   = ALLOC_B(NA * 6 * 256);     // CL [N,6,256], halo pad=1
  bf16*  x4B   = ALLOC_B(NA * 4 * 256);     // CL [N,4,256], no halo
  float* bnSums  = ALLOC_F(2 * 512);
  float* bnStats = ALLOC_F(2 * 512);

  // ---- pack weights ----
  pack_wt(stream, (const float*)d_in[P_NODE_W], node_wt, 92, 96, 64);
  pack_wt(stream, (const float*)d_in[P_EDGE_W], edge_wt, 41, 64, 64);
  for (int l = 0; l < 3; ++l) {
    const int cb = P_CONV0 + l * 14;
    pack_wt(stream, (const float*)d_in[cb + 3], pe_wt[l][0], 192, 192, 64);
    pack_wt(stream, (const float*)d_in[cb + 5], pe_wt[l][1], 64, 64, 64);
    pack_wt(stream, (const float*)d_in[cb + 7], pe_wt[l][2], 64, 64, 64);
    pack_wt(stream, (const float*)d_in[cb + 9], pv_wt[l][0], 128, 128, 64);
    pack_wt(stream, (const float*)d_in[cb + 11], pv_wt[l][1], 64, 64, 64);
    pack_wt(stream, (const float*)d_in[cb + 13], pv_wt[l][2], 64, 64, 64);
  }
  pack_wt(stream, (const float*)d_in[P_FEAT_W], feat_wt, 256, 256, 512);
  pack_wt(stream, (const float*)d_in[P_FD_W],   fd_wt, 128, 128, 128);
  pack_wt(stream, (const float*)d_in[P_ANF_W],  anf_wt, 128, 128, 128);
  pack_wt(stream, (const float*)d_in[P_CELL_W], cell_wt, 9, 32, 128);
  pack_wt(stream, (const float*)d_in[P_PP_W],   pp_wt, 896, 896, 512);
  pack_cw(stream, (const float*)d_in[P_C1_W], c1_wt, 512, 64, 3);
  pack_cw(stream, (const float*)d_in[P_C2_W], c2_wt, 512, 512, 3);
  pack_cw(stream, (const float*)d_in[P_C3_W], c3_wt, 256, 512, 3);
  pack_cw(stream, (const float*)d_in[P_C4_W], c4_wt, 256, 256, 3);

  // ---- embeddings ----
  cast_pad_kernel<<<(unsigned)((NA * 96 + 255) / 256), 256, 0, stream>>>(
      atom_fea, afB, NA, 92, 96);
  cast_pad_kernel<<<(unsigned)((NE * 64 + 255) / 256), 256, 0, stream>>>(
      nbr_fea, nfB, NE, 41, 64);
  gemm(stream, afB, 96, node_wt, (const float*)d_in[P_NODE_B], NA, 96, 64,
       atomF, 64, atomB, 64, 0);
  gemm(stream, nfB, 64, edge_wt, (const float*)d_in[P_EDGE_B], NE, 64, 64,
       nbrF, 64, nbrB, 64, 0);

  const long EA = NE * 64;
  const long NAF = NA * 64;
  const unsigned gEA = (unsigned)((EA + 255) / 256);
  const unsigned gNA = (unsigned)((NAF + 255) / 256);

  // ---- 3 message-passing layers ----
  for (int l = 0; l < 3; ++l) {
    const int cb = P_CONV0 + l * 14;
    const float* bn_b = (const float*)d_in[cb + 0];
    const float* bn_g = (const float*)d_in[cb + 1];

    edge_concat_kernel<<<(unsigned)((NE * 16 + 255) / 256), 256, 0, stream>>>(
        atomB, nbrB, idx1, idx2, Xe, NE);
    gemm(stream, Xe, 192, pe_wt[l][0], (const float*)d_in[cb + 2], NE, 192, 64,
         nullptr, 0, h1B, 64, 1);
    gemm(stream, h1B, 64, pe_wt[l][1], (const float*)d_in[cb + 4], NE, 64, 64,
         nullptr, 0, h2B, 64, 1);
    gemm(stream, h2B, 64, pe_wt[l][2], (const float*)d_in[cb + 6], NE, 64, 64,
         ekF, 64, nullptr, 0, 0);

    hipMemsetAsync(rho, 0, (size_t)NAF * 4, stream);
    scatter_scaled_kernel<<<gEA, 256, 0, stream>>>(ekF, idx1, num_nbrs, rho,
                                                   NE);
    concat2_cast_kernel<<<(unsigned)((NA * 128 + 255) / 256), 256, 0, stream>>>(
        atomF, rho, avB, NA, 64);
    gemm(stream, avB, 128, pv_wt[l][0], (const float*)d_in[cb + 8], NA, 128,
         64, nullptr, 0, hv1B, 64, 1);
    gemm(stream, hv1B, 64, pv_wt[l][1], (const float*)d_in[cb + 10], NA, 64,
         64, nullptr, 0, hv2B, 64, 1);
    gemm(stream, hv2B, 64, pv_wt[l][2], (const float*)d_in[cb + 12], NA, 64,
         64, viF, 64, nullptr, 0, 0);

    hipMemsetAsync(bnSums, 0, 2 * 64 * 4, stream);
    bn_stats_rows_kernel<<<768, 256, 2 * 64 * 4, stream>>>(viF, NA, 64,
                                                           bnSums);
    bn_finalize_kernel<<<1, 64, 0, stream>>>(bnSums, bnStats, 64, (float)NA);
    bn_res_update_kernel<<<gNA, 256, 0, stream>>>(atomF, atomB, viF, bnStats,
                                                  bn_g, bn_b, NA, 64);
    add_cast_kernel<<<gEA, 256, 0, stream>>>(nbrF, nbrB, ekF, EA);

    if (l == 2) {
      hipMemsetAsync(eksum, 0, (size_t)NAF * 4, stream);
      scatter_scaled_kernel<<<gEA, 256, 0, stream>>>(nbrF, idx1, num_nbrs,
                                                     eksum, NE);
      concat2_cast_kernel<<<(unsigned)((NA * 128 + 255) / 256), 256, 0,
                            stream>>>(atomF, eksum, anfInB, NA, 64);
    }
  }

  // ---- per-atom head ----
  head_gather_kernel<<<(unsigned)((NA + 255) / 256), 256, 0, stream>>>(
      crys, u_feat, r_feat, cell, urB, fdB, cellB, (int)NA);

  // ff = [anf(128) | ucell(128) | feature(512) | fdelta(128)] -> bf16 directly
  gemm(stream, anfInB, 128, anf_wt, (const float*)d_in[P_ANF_B], NA, 128, 128,
       nullptr, 0, ffB + 0, 896, 0);
  gemm(stream, cellB, 32, cell_wt, (const float*)d_in[P_CELL_B], NA, 32, 128,
       nullptr, 0, ffB + 128, 896, 0);
  gemm(stream, urB, 256, feat_wt, (const float*)d_in[P_FEAT_B], NA, 256, 512,
       nullptr, 0, ffB + 256, 896, 0);
  gemm(stream, fdB, 128, fd_wt, (const float*)d_in[P_FD_B], NA, 128, 128,
       nullptr, 0, ffB + 768, 896, 0);

  // phi_pos + BN + lrelu -> x0 channels-last [N,8,64]
  gemm(stream, ffB, 896, pp_wt, (const float*)d_in[P_PP_B], NA, 896, 512, pdF,
       512, nullptr, 0, 0);
  hipMemsetAsync(bnSums, 0, 2 * 512 * 4, stream);
  bn_stats_rows_kernel<<<768, 256, 2 * 512 * 4, stream>>>(pdF, NA, 512,
                                                          bnSums);
  bn_finalize_kernel<<<2, 256, 0, stream>>>(bnSums, bnStats, 512, (float)NA);
  pd_to_x0_kernel<<<(unsigned)((NA * 512 + 255) / 256), 256, 0, stream>>>(
      pdF, bnStats, (const float*)d_in[P_PP_BNG], (const float*)d_in[P_PP_BNB],
      x0B, NA);

  // c1: [N,8,64] -> y1 rows [N*6,512]; BN + lrelu -> x1 CL [N,6,512]
  conv(stream, x0B, 64, 8, c1_wt, (const float*)d_in[P_C1_B], NA, 6, 3, 512,
       y1F, nullptr, 0, 0, 0);
  hipMemsetAsync(bnSums, 0, 2 * 512 * 4, stream);
  bn_stats_rows_kernel<<<768, 256, 2 * 512 * 4, stream>>>(y1F, NA * 6, 512,
                                                          bnSums);
  bn_finalize_kernel<<<2, 256, 0, stream>>>(bnSums, bnStats, 512,
                                            (float)(NA * 6));
  bn_lrelu_cast_pad_kernel<<<(unsigned)((NA * 6 * 512 + 255) / 256), 256, 0,
                             stream>>>(y1F, bnStats,
                                       (const float*)d_in[P_C1_BNG],
                                       (const float*)d_in[P_C1_BNB], x1B, NA,
                                       6, 512, 6, 0);

  // c2: [N,6,512] -> y2 rows [N*4,512]; BN + lrelu -> x2 CL [N,6,512] pad=1
  conv(stream, x1B, 512, 6, c2_wt, (const float*)d_in[P_C2_B], NA, 4, 3, 512,
       y2F, nullptr, 0, 0, 0);
  hipMemsetAsync(bnSums, 0, 2 * 512 * 4, stream);
  bn_stats_rows_kernel<<<768, 256, 2 * 512 * 4, stream>>>(y2F, NA * 4, 512,
                                                          bnSums);
  bn_finalize_kernel<<<2, 256, 0, stream>>>(bnSums, bnStats, 512,
                                            (float)(NA * 4));
  zero_halo_kernel<<<(unsigned)((NA * 2 * 512 + 255) / 256), 256, 0, stream>>>(
      x2B, NA, 6, 512);
  bn_lrelu_cast_pad_kernel<<<(unsigned)((NA * 4 * 512 + 255) / 256), 256, 0,
                             stream>>>(y2F, bnStats,
                                       (const float*)d_in[P_C2_BNG],
                                       (const float*)d_in[P_C2_BNB], x2B, NA,
                                       4, 512, 6, 1);

  // c3 (pad=1, lrelu fused) -> x3 CL [N,6,256] interior, halo zeroed
  zero_halo_kernel<<<(unsigned)((NA * 2 * 256 + 255) / 256), 256, 0, stream>>>(
      x3B, NA, 6, 256);
  conv(stream, x2B, 512, 6, c3_wt, (const float*)d_in[P_C3_B], NA, 4, 3, 256,
       nullptr, x3B, 6, 1, 1);

  // c4 (pad=1, lrelu fused) -> x4 CL [N,4,256]
  conv(stream, x3B, 256, 6, c4_wt, (const float*)d_in[P_C4_B], NA, 4, 3, 256,
       nullptr, x4B, 4, 0, 1);

  // c5 -> out [N,3]
  c5_kernel<<<(unsigned)((NA * 3 + 255) / 256), 256, 0, stream>>>(
      x4B, (const float*)d_in[P_C5_W], (const float*)d_in[P_C5_B], out,
      (int)NA);
}
